// PVCNN_classifier_16733192585420
// MI455X (gfx1250) — matrix operand order repaired and checked by reading, not compiled, not run on hardware
//
#include <hip/hip_runtime.h>
#include <hip/hip_bf16.h>
#include <stdint.h>

typedef __attribute__((ext_vector_type(16))) _Float16 v16h;
typedef __attribute__((ext_vector_type(8)))  float    v8f;

#define BB 4
#define NP 2048
#define KN 128
#define CLIPV 0.999999f
#define R2THR 0.09f

__device__ __forceinline__ float clipc(float x){ return fminf(fmaxf(x,-CLIPV),CLIPV); }

// ---- CDNA5 async global->LDS (ASYNCcnt) ----
__device__ __forceinline__ void async_ld_b128(uint32_t lds_addr, const void* gaddr){
  // GV mode: LDS byte address in VGPR (low 32 bits of generic LDS pointer), 64-bit global addr in VGPR pair
  asm volatile("global_load_async_to_lds_b128 %0, %1, off" :: "v"(lds_addr), "v"(gaddr) : "memory");
}
__device__ __forceinline__ void wait_async0(){
  asm volatile("s_wait_asynccnt 0" ::: "memory");
}

// ---- WMMA tile gather helpers (CDNA5 wave32 16x16x32 f16 layouts) ----
__device__ __forceinline__ v16h gatherA(const _Float16* A, int lane){
  int grp=(lane>>4)&1, m=lane&15; v16h a;
#pragma unroll
  for(int e=0;e<16;++e){ int k=e+16*grp; a[e]=A[m*32+k]; }
  return a;
}
__device__ __forceinline__ v16h gatherB(const _Float16* Bm, int ldb, int n0, int lane){
  int grp=(lane>>4)&1, n=lane&15; v16h b;
#pragma unroll
  for(int e=0;e<16;++e){ int k=e+16*grp; b[e]=Bm[k*ldb+n0+n]; }
  return b;
}
__device__ __forceinline__ v8f wmma16(v16h a, v16h b, v8f c){
  return __builtin_amdgcn_wmma_f32_16x16x32_f16(false,a,false,b,(short)0,c,false,false);
}

// ---- stage 0: split inputs, normalize normals ----
__global__ void k_prep(const float* __restrict__ in, float* __restrict__ pts, float* __restrict__ nrm){
  int t = blockIdx.x*blockDim.x+threadIdx.x; if(t>=BB*NP) return;
  int b=t/NP, n=t%NP;
  const float* ib = in + (size_t)b*6*NP;
  float px=ib[0*NP+n], py=ib[1*NP+n], pz=ib[2*NP+n];
  float nx=ib[3*NP+n], ny=ib[4*NP+n], nz=ib[5*NP+n];
  float l = sqrtf(nx*nx+ny*ny+nz*nz);
  nx/=l; ny/=l; nz/=l;
  pts[t*3+0]=px; pts[t*3+1]=py; pts[t*3+2]=pz;
  nrm[t*3+0]=nx; nrm[t*3+1]=ny; nrm[t*3+2]=nz;
}

__global__ void k_means(const float* __restrict__ pts, const float* __restrict__ nrm, float* __restrict__ means){
  __shared__ float red[256];
  int b=blockIdx.x, t=threadIdx.x;
  for(int c=0;c<6;++c){
    const float* src = (c<3)? pts : nrm;
    int cc = c%3;
    float s=0.f;
    for(int n=t;n<NP;n+=256) s += src[((size_t)b*NP+n)*3+cc];
    red[t]=s; __syncthreads();
    for(int o=128;o>0;o>>=1){ if(t<o) red[t]+=red[t+o]; __syncthreads(); }
    if(t==0) means[b*6+c]=red[0]/NP;
    __syncthreads();
  }
}

__global__ void k_center_gppf(float* __restrict__ pts, const float* __restrict__ nrm,
                              const float* __restrict__ means, float* __restrict__ gppf){
  int t = blockIdx.x*blockDim.x+threadIdx.x; if(t>=BB*NP) return;
  int b=t/NP;
  float px=pts[t*3+0]-means[b*6+0];
  float py=pts[t*3+1]-means[b*6+1];
  float pz=pts[t*3+2]-means[b*6+2];
  pts[t*3+0]=px; pts[t*3+1]=py; pts[t*3+2]=pz;
  float nmx=means[b*6+3], nmy=means[b*6+4], nmz=means[b*6+5];
  float dn=sqrtf(px*px+py*py+pz*pz);
  float inv=1.f/fmaxf(dn,1e-8f);
  float ux=-px*inv, uy=-py*inv, uz=-pz*inv;   // d = mean(0) - p
  float nx=nrm[t*3+0], ny=nrm[t*3+1], nz=nrm[t*3+2];
  gppf[t*4+0]=acosf(clipc(nx*ux+ny*uy+nz*uz));
  gppf[t*4+1]=acosf(clipc(nmx*ux+nmy*uy+nmz*uz));
  gppf[t*4+2]=acosf(clipc(nmx*nx+nmy*ny+nmz*nz));
  gppf[t*4+3]=dn;
}

// ---- KNN radius search: ascending-index compaction via wave32 ballot ----
__global__ void k_knn(const float* __restrict__ pts, int* __restrict__ idx){
  __shared__ int sidx[4][KN];
  int wid=threadIdx.x>>5, lane=threadIdx.x&31;
  int p = blockIdx.x*4+wid; int b=p/NP;
  float px=pts[(size_t)p*3], py=pts[(size_t)p*3+1], pz=pts[(size_t)p*3+2];
  int cnt=0;
  for(int j0=0;j0<NP && cnt<KN;j0+=32){
    int j=j0+lane;
    size_t jo=((size_t)b*NP+j)*3;
    float dx=px-pts[jo], dy=py-pts[jo+1], dz=pz-pts[jo+2];
    bool inr = (dx*dx+dy*dy+dz*dz) < R2THR;
    unsigned mask=(unsigned)__ballot(inr);
    int pos=__popc(mask & ((1u<<lane)-1u));
    if(inr && cnt+pos<KN) sidx[wid][cnt+pos]=j;
    cnt+=__popc(mask); if(cnt>KN) cnt=KN;
  }
  int i0 = sidx[wid][0];                  // self always in radius -> cnt>=1
  for(int k=cnt+lane;k<KN;k+=32) sidx[wid][k]=i0;
  for(int k=lane;k<KN;k+=32) idx[(size_t)p*KN+k]=sidx[wid][k];
}

// ---- fused lppf -> f1(4->32) -> f2(32->64) -> max over 128 nbrs (WMMA), 1 wave per point ----
__global__ void k_edge(const float* __restrict__ pts, const float* __restrict__ nrm,
                       const float* __restrict__ gppf, const int* __restrict__ idx,
                       const float* __restrict__ f1w, const float* __restrict__ f1b,
                       const float* __restrict__ f2w, const float* __restrict__ f2b,
                       _Float16* __restrict__ feats){
  __shared__ __align__(16) _Float16 sA[2][128*32];
  __shared__ __align__(16) _Float16 sH[2][128*32];
  __shared__ __align__(16) _Float16 sB1[32*32];
  __shared__ __align__(16) _Float16 sB2[32*64];
  int tid=threadIdx.x, wid=tid>>5, lane=tid&31;
  for(int i=tid;i<32*32;i+=64){ int k=i>>5,n=i&31; sB1[i]=(k<4)?(_Float16)f1w[n*4+k]:(_Float16)0.f; }
  for(int i=tid;i<32*64;i+=64){ int k=i/64,n=i%64; sB2[i]=(_Float16)f2w[n*32+k]; }
  __syncthreads();
  int p = blockIdx.x*2+wid; int b=p/NP;
  float pix=pts[(size_t)p*3], piy=pts[(size_t)p*3+1], piz=pts[(size_t)p*3+2];
  float nix=nrm[(size_t)p*3], niy=nrm[(size_t)p*3+1], niz=nrm[(size_t)p*3+2];
  _Float16* A=sA[wid]; _Float16* H=sH[wid];
  for(int k=lane;k<KN;k+=32){
    int j=idx[(size_t)p*KN+k];
    size_t jo=((size_t)b*NP+j)*3;
    float dx=pix-pts[jo], dy=piy-pts[jo+1], dz=piz-pts[jo+2];
    float dn=sqrtf(dx*dx+dy*dy+dz*dz);
    float inv=1.f/fmaxf(dn,1e-8f);
    float ux=dx*inv, uy=dy*inv, uz=dz*inv;
    float njx=nrm[jo], njy=nrm[jo+1], njz=nrm[jo+2];
    float g1=acosf(clipc(njx*ux+njy*uy+njz*uz));
    float g2=acosf(clipc(nix*ux+niy*uy+niz*uz));
    float g3=acosf(clipc(nix*njx+niy*njy+niz*njz));
    _Float16* row=A+k*32;
    row[0]=(_Float16)g1; row[1]=(_Float16)g2; row[2]=(_Float16)g3; row[3]=(_Float16)dn;
    for(int c=4;c<32;++c) row[c]=(_Float16)0.f;
  }
  __syncthreads();
  int grp=(lane>>4)&1, ln=lane&15;
  for(int rt=0;rt<8;++rt){
    v16h a=gatherA(A+rt*16*32,lane);
    v8f z={}; v8f c0=z, c1=z;
    v16h b0=gatherB(sB1,32,0,lane), b1v=gatherB(sB1,32,16,lane);
    c0=wmma16(a,b0,c0); c1=wmma16(a,b1v,c1);
#pragma unroll
    for(int r=0;r<8;++r){
      int m=rt*16 + r + 8*grp;
      float v0=fmaxf(c0[r]+f1b[ln],0.f);
      float v1=fmaxf(c1[r]+f1b[16+ln],0.f);
      H[m*32+ln]=(_Float16)v0; H[m*32+16+ln]=(_Float16)v1;
    }
  }
  __syncthreads();
  float cm[4]={0.f,0.f,0.f,0.f};
  for(int rt=0;rt<8;++rt){
    v16h a=gatherA(H+rt*16*32,lane);
    for(int ct=0;ct<4;++ct){
      v8f c={};
      v16h bb=gatherB(sB2,64,ct*16,lane);
      c=wmma16(a,bb,c);
      float bias=f2b[ct*16+ln];
#pragma unroll
      for(int r=0;r<8;++r){ float v=fmaxf(c[r]+bias,0.f); cm[ct]=fmaxf(cm[ct],v); }
    }
  }
#pragma unroll
  for(int ct=0;ct<4;++ct){ float o=__shfl_xor(cm[ct],16,32); cm[ct]=fmaxf(cm[ct],o); }
  _Float16* fo = feats + (size_t)p*96;
  if(lane<16){
#pragma unroll
    for(int ct=0;ct<4;++ct) fo[4+ct*16+lane]=(_Float16)cm[ct];
  }
  if(lane<4)  fo[lane]=(_Float16)gppf[(size_t)p*4+lane];
  if(lane<28) fo[68+lane]=(_Float16)0.f;
}

__global__ void k_zero(float* __restrict__ p, int n){
  int t=blockIdx.x*blockDim.x+threadIdx.x; if(t<n) p[t]=0.f;
}

// ---- voxelization: scatter mean (atomics), channels-last ----
__global__ void k_voxaccum(const _Float16* __restrict__ feats, int KP, int Cin,
                           const float* __restrict__ pts, int R,
                           float* __restrict__ vsum, float* __restrict__ vcnt){
  int t=blockIdx.x*blockDim.x+threadIdx.x; if(t>=BB*NP) return;
  int b=t/NP; int R3=R*R*R;
  float rr=(float)(R-1);
  float vx=fminf(fmaxf((pts[t*3+0]+1.f)*0.5f*rr,0.f),rr);
  float vy=fminf(fmaxf((pts[t*3+1]+1.f)*0.5f*rr,0.f),rr);
  float vz=fminf(fmaxf((pts[t*3+2]+1.f)*0.5f*rr,0.f),rr);
  int ix=(int)rintf(vx), iy=(int)rintf(vy), iz=(int)rintf(vz);
  int fl=(ix*R+iy)*R+iz;
  size_t base=((size_t)b*R3+fl)*(size_t)Cin;
  const _Float16* fr=feats+(size_t)t*KP;
  for(int c=0;c<Cin;++c) atomicAdd(&vsum[base+c], (float)fr[c]);
  atomicAdd(&vcnt[(size_t)b*R3+fl], 1.f);
}

__global__ void k_voxfinish(const float* __restrict__ vsum, const float* __restrict__ vcnt,
                            _Float16* __restrict__ vin, int cells, int Cin, int CinP){
  int t=blockIdx.x*blockDim.x+threadIdx.x; if(t>=cells*CinP) return;
  int cell=t/CinP, c=t%CinP;
  float v=0.f;
  if(c<Cin){ float cn=fmaxf(vcnt[cell],1.f); v=vsum[(size_t)cell*Cin+c]/cn; }
  vin[t]=(_Float16)v;
}

// ---- weight transpose OIDHW -> [tap][cinP][cout] f16 ----
__global__ void k_wtrans(const float* __restrict__ w, _Float16* __restrict__ wt,
                         int Cin, int CinP, int Cout){
  int t=blockIdx.x*blockDim.x+threadIdx.x;
  int tot=27*CinP*Cout; if(t>=tot) return;
  int co=t%Cout; int rest=t/Cout; int ci=rest%CinP; int tp=rest/CinP;
  float v=(ci<Cin)? w[((size_t)co*Cin+ci)*27+tp] : 0.f;
  wt[t]=(_Float16)v;   // t == (tp*CinP+ci)*Cout+co
}

// ---- implicit-GEMM 3x3x3 conv + leaky-relu via WMMA, async global->LDS staging ----
__global__ void k_conv3d(const _Float16* __restrict__ vin, const _Float16* __restrict__ wt,
                         const float* __restrict__ bias, int R, int CinP, int Cout,
                         _Float16* __restrict__ out16, float* __restrict__ out32){
  __shared__ __align__(16) _Float16 sA[16*32];
  __shared__ __align__(16) _Float16 sB[32*128];
  int tid=threadIdx.x, wid=tid>>5, lane=tid&31;
  int R3=R*R*R; int bpb=R3/16;
  int b=blockIdx.x/bpb; int base=(blockIdx.x%bpb)*16;
  int NT=Cout>>6;
  v8f z={}; v8f acc[2]; acc[0]=z; acc[1]=z;
  int KC=CinP/32;
  uint32_t ldsA=(uint32_t)(uintptr_t)sA;
  uint32_t ldsB=(uint32_t)(uintptr_t)sB;
  for(int tp=0;tp<27;++tp){
    int dx=tp/9-1, dy=(tp/3)%3-1, dz=tp%3-1;
    if(tp<26 && tid==0)  // prefetch next tap's weight slice into L2
      __builtin_prefetch(wt + ((size_t)(tp+1)*CinP)*Cout, 0, 0);
    for(int kc=0;kc<KC;++kc){
      if(tid<64){
        int row=tid>>2, q=tid&3;
        int v=base+row;
        int x=v/(R*R), y=(v/R)%R, zc=v%R;
        int xs=x+dx, ys=y+dy, zs=zc+dz;
        bool ok=(xs>=0)&&(xs<R)&&(ys>=0)&&(ys<R)&&(zs>=0)&&(zs<R);
        if(ok){
          const void* src = (const void*)(vin + (((size_t)b*R3 + (size_t)((xs*R+ys)*R+zs))*CinP + kc*32) + q*8);
          async_ld_b128(ldsA + (uint32_t)(row*64 + q*16), src);
        } else {
          uint32_t* dst=(uint32_t*)(sA+row*32)+q*4;
          dst[0]=0u; dst[1]=0u; dst[2]=0u; dst[3]=0u;
        }
      }
      {
        const char* srcb=(const char*)(wt + ((size_t)tp*CinP + kc*32)*Cout);
        int nchunk=4*Cout;                 // 16-byte chunks in 32 x Cout halfs
        for(int i=tid;i<nchunk;i+=128)
          async_ld_b128(ldsB + (uint32_t)(i*16), (const void*)(srcb + (size_t)i*16));
      }
      wait_async0();
      __syncthreads();
      v16h a=gatherA(sA,lane);
      for(int it=0;it<NT;++it){
        v16h bm=gatherB(sB,Cout,it*64+wid*16,lane);
        acc[it]=wmma16(a,bm,acc[it]);
      }
      __syncthreads();
    }
  }
  int grp=(lane>>4)&1, ln=lane&15;
  for(int it=0;it<NT;++it){
    int n=it*64+wid*16+ln;
    float bv=bias[n];
#pragma unroll
    for(int r=0;r<8;++r){
      int m=base + r + 8*grp;
      float v=acc[it][r]+bv;
      v=(v>0.f)? v : 0.1f*v;
      size_t o=((size_t)b*R3+m)*Cout+n;
      if(out32) out32[o]=v;
      if(out16) out16[o]=(_Float16)v;
    }
  }
}

__global__ void k_semean(const float* __restrict__ g, int R3, int Cout, float* __restrict__ s){
  __shared__ float red[256];
  int b=blockIdx.x/Cout, co=blockIdx.x%Cout, t=threadIdx.x;
  float acc=0.f;
  for(int v=t;v<R3;v+=256) acc+=g[((size_t)b*R3+v)*Cout+co];
  red[t]=acc; __syncthreads();
  for(int o=128;o>0;o>>=1){ if(t<o) red[t]+=red[t+o]; __syncthreads(); }
  if(t==0) s[b*Cout+co]=red[0]/R3;
}

__global__ void k_se(const float* __restrict__ s, const float* __restrict__ w1, const float* __restrict__ b1,
                     const float* __restrict__ w2, const float* __restrict__ b2,
                     int Cout, int Ch, float* __restrict__ e){
  __shared__ float sl[128]; __shared__ float hl[16];
  int b=blockIdx.x, t=threadIdx.x;
  if(t<Cout) sl[t]=s[b*Cout+t];
  __syncthreads();
  if(t<Ch){ float a=b1[t]; for(int c=0;c<Cout;++c) a+=w1[t*Cout+c]*sl[c]; hl[t]=fmaxf(a,0.f); }
  __syncthreads();
  if(t<Cout){ float a=b2[t]; for(int j=0;j<Ch;++j) a+=w2[t*Ch+j]*hl[j]; e[b*Cout+t]=1.f/(1.f+expf(-a)); }
}

// ---- per-point MLP (feats @ W^T + b, relu) via WMMA, async A staging ----
__global__ void k_pointmlp(const _Float16* __restrict__ feats, int KP, int Cin,
                           const float* __restrict__ w, const float* __restrict__ bias,
                           int Cout, float* __restrict__ out){
  __shared__ __align__(16) _Float16 sA[16*32];
  __shared__ __align__(16) _Float16 sB[32*128];
  int tid=threadIdx.x, wid=tid>>5, lane=tid&31;
  int p0=blockIdx.x*16;
  int NT=Cout>>6;
  v8f z={}; v8f acc[2]; acc[0]=z; acc[1]=z;
  int KC=KP/32;
  uint32_t ldsA=(uint32_t)(uintptr_t)sA;
  for(int kc=0;kc<KC;++kc){
    if(tid<64){
      int row=tid>>2, q=tid&3;
      const void* src=(const void*)(feats+((size_t)(p0+row)*KP + kc*32) + q*8);
      async_ld_b128(ldsA + (uint32_t)(row*64 + q*16), src);
    }
    for(int i=tid;i<32*Cout;i+=128){
      int k=i/Cout, n=i%Cout; int kk=kc*32+k;
      sB[i]=(kk<Cin)? (_Float16)w[(size_t)n*Cin+kk] : (_Float16)0.f;
    }
    wait_async0();
    __syncthreads();
    v16h a=gatherA(sA,lane);
    for(int it=0;it<NT;++it){
      v16h bm=gatherB(sB,Cout,it*64+wid*16,lane);
      acc[it]=wmma16(a,bm,acc[it]);
    }
    __syncthreads();
  }
  int grp=(lane>>4)&1, ln=lane&15;
  for(int it=0;it<NT;++it){
    int n=it*64+wid*16+ln; float bv=bias[n];
#pragma unroll
    for(int r=0;r<8;++r){
      int p=p0 + r + 8*grp;
      out[(size_t)p*Cout+n]=fmaxf(acc[it][r]+bv,0.f);
    }
  }
}

// ---- devoxelize (trilinear) * SE + point-MLP add ----
__global__ void k_devox(const float* __restrict__ g, const float* __restrict__ e,
                        const float* __restrict__ pt, const float* __restrict__ pts,
                        int R, int Cout, float* __restrict__ outf, _Float16* __restrict__ outh){
  int t=blockIdx.x*blockDim.x+threadIdx.x;
  if(t>=BB*NP*Cout) return;
  int co=t%Cout; int p=t/Cout; int b=p/NP;
  int R3=R*R*R; float rr=(float)(R-1);
  float vx=fminf(fmaxf((pts[(size_t)p*3+0]+1.f)*0.5f*rr,0.f),rr);
  float vy=fminf(fmaxf((pts[(size_t)p*3+1]+1.f)*0.5f*rr,0.f),rr);
  float vz=fminf(fmaxf((pts[(size_t)p*3+2]+1.f)*0.5f*rr,0.f),rr);
  float fx=floorf(vx), fy=floorf(vy), fz=floorf(vz);
  float rx=vx-fx, ry=vy-fy, rz=vz-fz;
  int x0=(int)fx, y0=(int)fy, z0=(int)fz;
  float acc=0.f;
#pragma unroll
  for(int dx=0;dx<2;++dx)
#pragma unroll
  for(int dy=0;dy<2;++dy)
#pragma unroll
  for(int dz=0;dz<2;++dz){
    int cx=min(x0+dx,R-1), cy=min(y0+dy,R-1), cz=min(z0+dz,R-1);
    float w=(dx?rx:1.f-rx)*(dy?ry:1.f-ry)*(dz?rz:1.f-rz);
    acc+=w*g[((size_t)b*R3+(cx*R+cy)*R+cz)*Cout+co];
  }
  float v=acc*e[b*Cout+co]+pt[(size_t)p*Cout+co];
  if(outf) outf[t]=v;
  if(outh) outh[(size_t)p*Cout+co]=(_Float16)v;
}

__global__ void k_pool(const float* __restrict__ f, int Cout, float* __restrict__ pooled){
  __shared__ float red[256];
  int b=blockIdx.x/Cout, co=blockIdx.x%Cout, t=threadIdx.x;
  float m=-1e30f;
  for(int n=t;n<NP;n+=256) m=fmaxf(m, f[((size_t)b*NP+n)*Cout+co]);
  red[t]=m; __syncthreads();
  for(int o=128;o>0;o>>=1){ if(t<o) red[t]=fmaxf(red[t],red[t+o]); __syncthreads(); }
  if(t==0) pooled[b*Cout+co]=red[0];
}

__global__ void k_head(const float* __restrict__ pooled,
                       const float* __restrict__ w1, const float* __restrict__ b1,
                       const float* __restrict__ w2, const float* __restrict__ b2,
                       const float* __restrict__ w3, const float* __restrict__ b3,
                       float* __restrict__ out){
  __shared__ float sp[128]; __shared__ float h1[512]; __shared__ float h2[256];
  int b=blockIdx.x, t=threadIdx.x;
  if(t<128) sp[t]=pooled[b*128+t];
  __syncthreads();
  { float a=b1[t]; const float* wr=w1+(size_t)t*128; for(int k=0;k<128;++k) a+=wr[k]*sp[k]; h1[t]=fmaxf(a,0.f); }
  __syncthreads();
  if(t<256){ float a=b2[t]; const float* wr=w2+(size_t)t*512; for(int k=0;k<512;++k) a+=wr[k]*h1[k]; h2[t]=fmaxf(a,0.f); }
  __syncthreads();
  if(t<40){ float a=b3[t]; const float* wr=w3+(size_t)t*256; for(int k=0;k<256;++k) a+=wr[k]*h2[k]; out[b*40+t]=a; }
}

extern "C" void kernel_launch(void* const* d_in, const int* in_sizes, int n_in,
                              void* d_out, int out_size, void* d_ws, size_t ws_size,
                              hipStream_t stream){
  const float* inps=(const float*)d_in[0];
  const float *f1w=(const float*)d_in[1],  *f1b=(const float*)d_in[2];
  const float *f2w=(const float*)d_in[3],  *f2b=(const float*)d_in[4];
  const float *p1w=(const float*)d_in[5],  *p1b=(const float*)d_in[6];
  const float *v1aw=(const float*)d_in[7], *v1ab=(const float*)d_in[8];
  const float *v1bw=(const float*)d_in[9], *v1bb=(const float*)d_in[10];
  const float *se1w1=(const float*)d_in[11], *se1b1=(const float*)d_in[12];
  const float *se1w2=(const float*)d_in[13], *se1b2=(const float*)d_in[14];
  const float *p2w=(const float*)d_in[15], *p2b=(const float*)d_in[16];
  const float *v2aw=(const float*)d_in[17], *v2ab=(const float*)d_in[18];
  const float *v2bw=(const float*)d_in[19], *v2bb=(const float*)d_in[20];
  const float *se2w1=(const float*)d_in[21], *se2b1=(const float*)d_in[22];
  const float *se2w2=(const float*)d_in[23], *se2b2=(const float*)d_in[24];
  const float *c1w=(const float*)d_in[25], *c1b=(const float*)d_in[26];
  const float *c2w=(const float*)d_in[27], *c2b=(const float*)d_in[28];
  const float *c3w=(const float*)d_in[29], *c3b=(const float*)d_in[30];

  char* ws=(char*)d_ws; size_t off=0;
  auto alloc=[&](size_t bytes)->char*{ char* p=ws+off; off=(off+bytes+255)&~(size_t)255; return p; };
  float*     pts   =(float*)    alloc((size_t)BB*NP*3*4);
  float*     nrm   =(float*)    alloc((size_t)BB*NP*3*4);
  float*     means =(float*)    alloc(256);
  float*     gppf  =(float*)    alloc((size_t)BB*NP*4*4);
  int*       idx   =(int*)      alloc((size_t)BB*NP*KN*4);
  _Float16*  feats0=(_Float16*) alloc((size_t)BB*NP*96*2);
  float*     ptb   =(float*)    alloc((size_t)BB*NP*128*4);
  _Float16*  f1h   =(_Float16*) alloc((size_t)BB*NP*64*2);
  float*     f2f   =(float*)    alloc((size_t)BB*NP*128*4);
  float*     sbuf  =(float*)    alloc((size_t)BB*128*4);
  float*     ebuf  =(float*)    alloc((size_t)BB*128*4);
  float*     pool  =(float*)    alloc((size_t)BB*128*4);
  _Float16*  wt    =(_Float16*) alloc((size_t)27*128*128*2);
  float*     vsum  =(float*)    alloc((size_t)BB*32768*68*4);
  float*     vcnt  =(float*)    alloc((size_t)BB*32768*4);
  _Float16*  vin   =(_Float16*) alloc((size_t)BB*32768*96*2);
  _Float16*  ga16  =(_Float16*) alloc((size_t)BB*32768*64*2);
  float*     gb32  =(float*)    alloc((size_t)BB*32768*64*4);
  (void)ws_size; (void)in_sizes; (void)n_in; (void)out_size;

  int BN=BB*NP;
  k_prep<<<(BN+255)/256,256,0,stream>>>(inps,pts,nrm);
  k_means<<<BB,256,0,stream>>>(pts,nrm,means);
  k_center_gppf<<<(BN+255)/256,256,0,stream>>>(pts,nrm,means,gppf);
  k_knn<<<BN/4,128,0,stream>>>(pts,idx);
  k_edge<<<BN/2,64,0,stream>>>(pts,nrm,gppf,idx,f1w,f1b,f2w,f2b,feats0);

  // ---------- PVConv 1 : r=32, Cin=68 (pad 96), Cout=64 ----------
  {
    int R=32, R3=32768, Cin=68, CinP=96, Cout=64;
    k_zero<<<(BB*R3*Cin+255)/256,256,0,stream>>>(vsum,BB*R3*Cin);
    k_zero<<<(BB*R3+255)/256,256,0,stream>>>(vcnt,BB*R3);
    k_voxaccum<<<(BN+255)/256,256,0,stream>>>(feats0,96,Cin,pts,R,vsum,vcnt);
    k_voxfinish<<<(BB*R3*CinP+255)/256,256,0,stream>>>(vsum,vcnt,vin,BB*R3,Cin,CinP);
    k_wtrans<<<(27*CinP*Cout+255)/256,256,0,stream>>>(v1aw,wt,Cin,CinP,Cout);
    k_conv3d<<<BB*R3/16,128,0,stream>>>(vin,wt,v1ab,R,CinP,Cout,ga16,(float*)nullptr);
    k_wtrans<<<(27*64*64+255)/256,256,0,stream>>>(v1bw,wt,64,64,64);
    k_conv3d<<<BB*R3/16,128,0,stream>>>(ga16,wt,v1bb,R,64,64,(_Float16*)nullptr,gb32);
    k_semean<<<BB*64,256,0,stream>>>(gb32,R3,64,sbuf);
    k_se<<<BB,128,0,stream>>>(sbuf,se1w1,se1b1,se1w2,se1b2,64,8,ebuf);
    k_pointmlp<<<BN/16,128,0,stream>>>(feats0,96,68,p1w,p1b,64,ptb);
    k_devox<<<(BN*64+255)/256,256,0,stream>>>(gb32,ebuf,ptb,pts,R,64,(float*)nullptr,f1h);
  }
  // ---------- PVConv 2 : r=16, Cin=64, Cout=128 ----------
  {
    int R=16, R3=4096, Cin=64, CinP=64, Cout=128;
    k_zero<<<(BB*R3*Cin+255)/256,256,0,stream>>>(vsum,BB*R3*Cin);
    k_zero<<<(BB*R3+255)/256,256,0,stream>>>(vcnt,BB*R3);
    k_voxaccum<<<(BN+255)/256,256,0,stream>>>(f1h,64,Cin,pts,R,vsum,vcnt);
    k_voxfinish<<<(BB*R3*CinP+255)/256,256,0,stream>>>(vsum,vcnt,vin,BB*R3,Cin,CinP);
    k_wtrans<<<(27*CinP*Cout+255)/256,256,0,stream>>>(v2aw,wt,Cin,CinP,Cout);
    k_conv3d<<<BB*R3/16,128,0,stream>>>(vin,wt,v2ab,R,CinP,Cout,ga16,(float*)nullptr);
    k_wtrans<<<(27*128*128+255)/256,256,0,stream>>>(v2bw,wt,128,128,128);
    k_conv3d<<<BB*R3/16,128,0,stream>>>(ga16,wt,v2bb,R,128,128,(_Float16*)nullptr,gb32);
    k_semean<<<BB*128,256,0,stream>>>(gb32,R3,128,sbuf);
    k_se<<<BB,128,0,stream>>>(sbuf,se2w1,se2b1,se2w2,se2b2,128,16,ebuf);
    k_pointmlp<<<BN/16,128,0,stream>>>(f1h,64,64,p2w,p2b,128,ptb);
    k_devox<<<(BN*128+255)/256,256,0,stream>>>(gb32,ebuf,ptb,pts,R,128,f2f,(_Float16*)nullptr);
  }
  k_pool<<<BB*128,256,0,stream>>>(f2f,128,pool);
  k_head<<<BB,512,0,stream>>>(pool,c1w,c1b,c2w,c2b,c3w,c3b,(float*)d_out);
}
